// DecoderTransformerLayer_90864328114943
// MI455X (gfx1250) — matrix operand order repaired and checked by reading, not compiled, not run on hardware
//
#include <hip/hip_runtime.h>
#include <hip/hip_bf16.h>
#include <math.h>

// ---------------------------------------------------------------------------
// Problem constants (from reference)
// ---------------------------------------------------------------------------
constexpr int Bc   = 2;
constexpr int Sc   = 2048;
constexpr int Hc   = 16;
constexpr int Dc   = 64;
constexpr int HIDc = 512;
constexpr int INTERc = 1024;
constexpr float EPSf   = 1e-8f;
constexpr float SCALEf = 0.125f;           // D^-0.5 = 1/8

// ---------------------------------------------------------------------------
// Types / helpers
// ---------------------------------------------------------------------------
typedef __attribute__((ext_vector_type(16))) __bf16 v16bf;
typedef __attribute__((ext_vector_type(8)))  __bf16 v8bf;
typedef __attribute__((ext_vector_type(4)))  __bf16 v4bf;
typedef __attribute__((ext_vector_type(8)))  float  v8f;
typedef __attribute__((ext_vector_type(4)))  float  v4f;

__device__ __forceinline__ __bf16 f2bf(float f) {
    return (__bf16)f;                       // native v_cvt bf16 path
}
__device__ __forceinline__ v4bf cvt4(v4f x) {
    return __builtin_convertvector(x, v4bf); // -> packed bf16 converts
}

// build a 16xbf16 A/B fragment from two 16B-aligned LDS chunks
__device__ __forceinline__ v16bf ld_frag(const __bf16* lo, const __bf16* hi) {
    v8bf a = *(const v8bf*)lo;
    v8bf b = *(const v8bf*)hi;
    return __builtin_shufflevector(a, b, 0,1,2,3,4,5,6,7,8,9,10,11,12,13,14,15);
}

// ---------------------------------------------------------------------------
// RMSNorm:  out[row,:] = x[row,:] * rsqrt(mean(x^2)+eps) * w
// one 256-thread block per row of HID=512
// ---------------------------------------------------------------------------
__global__ __launch_bounds__(256)
void rmsnorm_kernel(const float* __restrict__ x, const float* __restrict__ w,
                    float* __restrict__ out) {
    __shared__ float red[256];
    const int row = blockIdx.x;
    const int t   = threadIdx.x;
    const float* xr = x + (size_t)row * HIDc;
    float v0 = xr[t];
    float v1 = xr[t + 256];
    red[t] = v0 * v0 + v1 * v1;
    __syncthreads();
    for (int st = 128; st > 0; st >>= 1) {
        if (t < st) red[t] += red[t + st];
        __syncthreads();
    }
    float scale = rsqrtf(red[0] * (1.0f / HIDc) + EPSf);
    float* orow = out + (size_t)row * HIDc;
    orow[t]       = v0 * scale * w[t];
    orow[t + 256] = v1 * scale * w[t + 256];
}

// ---------------------------------------------------------------------------
// bf16 WMMA GEMM:  C[M,N] = epi(A[M,K] @ B[K,N])
//   epi: plain store, or  C = res + cscale[col]*acc   (when cscale != null)
// 256 threads (8 waves), macro-tile 128x128, K-block 64 (two 32-wide wmma steps).
// wave (waveM 0..3, waveN 0..1) computes 32x64 = 2x4 WMMA tiles.
// ---------------------------------------------------------------------------
__global__ __launch_bounds__(256)
void gemm_bf16_kernel(const float* __restrict__ A, const float* __restrict__ Bm,
                      float* __restrict__ C, const float* __restrict__ res,
                      const float* __restrict__ cscale, int K, int N) {
    __shared__ __align__(16) __bf16 ldsA [128][64];   // [m][k]
    __shared__ __align__(16) __bf16 ldsBT[128][64];   // [n][k]  (B transposed)

    const int tid = threadIdx.x;
    const int w   = tid >> 5;
    const int l   = tid & 31;
    const int lm  = l & 15;          // lane % 16
    const int lh  = l >> 4;          // lane half (0/1)
    const int waveM = w >> 1, waveN = w & 1;
    const int blockM = blockIdx.y * 128, blockN = blockIdx.x * 128;

    v8f acc[2][4];
    for (int mi = 0; mi < 2; ++mi)
        for (int ni = 0; ni < 4; ++ni)
            for (int r = 0; r < 8; ++r) acc[mi][ni][r] = 0.0f;

    // staging maps (8192 elems each / 256 threads = 32 per thread):
    //   A: thread -> (row = t/2, 32-col half); B: thread -> (k = t/4, 32-n chunk)
    const int arow = tid >> 1,  ac0 = (tid & 1) * 32;
    const int bk   = tid >> 2,  bn0 = (tid & 3) * 32;

    for (int k0 = 0; k0 < K; k0 += 64) {
        const float* pA = A  + (size_t)(blockM + arow) * K + k0 + ac0;
        const float* pB = Bm + (size_t)(k0 + bk) * N + blockN + bn0;
        #pragma unroll
        for (int jj = 0; jj < 8; ++jj) {
            v4bf a4 = cvt4(*(const v4f*)(pA + jj * 4));
            *(v4bf*)&ldsA[arow][ac0 + jj * 4] = a4;
        }
        #pragma unroll
        for (int jj = 0; jj < 8; ++jj) {
            v4bf b4 = cvt4(*(const v4f*)(pB + jj * 4));
            #pragma unroll
            for (int e = 0; e < 4; ++e) ldsBT[bn0 + jj * 4 + e][bk] = b4[e];
        }
        if (k0 + 64 < K) {                       // -> global_prefetch_b8
            __builtin_prefetch(pA + 64, 0, 0);
            __builtin_prefetch(pB + (size_t)64 * N, 0, 0);
        }
        __syncthreads();

        #pragma unroll
        for (int s = 0; s < 2; ++s) {
            v16bf afr[2], bfr[4];
            #pragma unroll
            for (int mi = 0; mi < 2; ++mi) {
                const int m = waveM * 32 + mi * 16 + lm;
                afr[mi] = ld_frag(&ldsA[m][s * 32 + lh * 8],
                                  &ldsA[m][s * 32 + 16 + lh * 8]);
            }
            #pragma unroll
            for (int ni = 0; ni < 4; ++ni) {
                const int n = waveN * 64 + ni * 16 + lm;
                bfr[ni] = ld_frag(&ldsBT[n][s * 32 + lh * 8],
                                  &ldsBT[n][s * 32 + 16 + lh * 8]);
            }
            #pragma unroll
            for (int mi = 0; mi < 2; ++mi)
                #pragma unroll
                for (int ni = 0; ni < 4; ++ni)
                    acc[mi][ni] = __builtin_amdgcn_wmma_f32_16x16x32_bf16(
                        false, afr[mi], false, bfr[ni], (short)0, acc[mi][ni],
                        false, false);
        }
        __syncthreads();
    }

    // epilogue: C/D layout row = lh*8 + r, col = lm
    for (int mi = 0; mi < 2; ++mi)
        for (int ni = 0; ni < 4; ++ni)
            for (int r = 0; r < 8; ++r) {
                const int row = blockM + waveM * 32 + mi * 16 + lh * 8 + r;
                const int col = blockN + waveN * 64 + ni * 16 + lm;
                const size_t idx = (size_t)row * N + col;
                float val = acc[mi][ni][r];
                if (cscale) val = res[idx] + cscale[col] * val;
                C[idx] = val;
            }
}

// ---------------------------------------------------------------------------
// RoPE in place on q and k, layout [B,S,H,D]
// ---------------------------------------------------------------------------
__global__ __launch_bounds__(256)
void rope_kernel(float* __restrict__ q, float* __restrict__ k,
                 const float* __restrict__ cosv, const float* __restrict__ sinv) {
    const int t = blockIdx.x * 256 + threadIdx.x;    // B*S*H*32 total
    const int i = t & 31;
    const int h = (t >> 5) & (Hc - 1);
    const int s = (t >> 9) & (Sc - 1);
    const int b = t >> 20;
    if (b >= Bc) return;
    const float c  = cosv[s * 32 + i];
    const float sn = sinv[s * 32 + i];
    const size_t base = (((size_t)b * Sc + s) * Hc + h) * Dc;
    float q1 = q[base + i], q2 = q[base + 32 + i];
    q[base + i]      = q1 * c - q2 * sn;
    q[base + 32 + i] = q2 * c + q1 * sn;
    float k1 = k[base + i], k2 = k[base + 32 + i];
    k[base + i]      = k1 * c - k2 * sn;
    k[base + 32 + i] = k2 * c + k1 * sn;
}

// ---------------------------------------------------------------------------
// Flash attention, causal. grid = (B*H, S/64), 128 threads = 4 waves.
// Each wave owns 16 query rows; loop over 32-key chunks; QK^T and P@V via WMMA.
// q/k/v/out layout: [B,S,H,D] fp32.
// ---------------------------------------------------------------------------
__global__ __launch_bounds__(128)
void attn_kernel(const float* __restrict__ q, const float* __restrict__ k,
                 const float* __restrict__ v, float* __restrict__ o) {
    __shared__ __align__(16) __bf16 ldsK [32][64];      // [key][d]
    __shared__ __align__(16) __bf16 ldsVT[64][32];      // [d][key]
    __shared__ __align__(16) __bf16 ldsP [4][16][32];   // per-wave P tile

    const int bh = blockIdx.x;
    const int b  = bh / Hc, h = bh % Hc;
    const int qbase = blockIdx.y * 64;
    const int tid = threadIdx.x;
    const int w = tid >> 5, l = tid & 31;
    const int lm = l & 15, lh = l >> 4;
    const int qrow0 = qbase + w * 16;

    // Q fragments (A-layout), two K-steps of 32 over D=64
    v16bf aQ[2];
    {
        const float* qp = q + (((size_t)b * Sc + (qrow0 + lm)) * Hc + h) * Dc;
        #pragma unroll
        for (int s = 0; s < 2; ++s)
            #pragma unroll
            for (int i = 0; i < 8; ++i) {
                aQ[s][i]     = f2bf(qp[s * 32 + lh * 8 + i]);
                aQ[s][8 + i] = f2bf(qp[s * 32 + 16 + lh * 8 + i]);
            }
    }

    float rmax[8], rsum[8];
    v8f oacc[4];
    for (int r = 0; r < 8; ++r) { rmax[r] = -__builtin_inff(); rsum[r] = 0.0f; }
    for (int t2 = 0; t2 < 4; ++t2)
        for (int r = 0; r < 8; ++r) oacc[t2][r] = 0.0f;

    const int nChunks = (qbase + 64) / 32;
    for (int c = 0; c < nChunks; ++c) {
        const int kb = c * 32;
        // cooperative stage of K (row-major) and V (transposed), fp32 -> bf16
        {
            const int key = tid >> 2;
            const int d0  = (tid & 3) * 16;
            const float* kp = k + (((size_t)b * Sc + (kb + key)) * Hc + h) * Dc + d0;
            const float* vp = v + (((size_t)b * Sc + (kb + key)) * Hc + h) * Dc + d0;
            #pragma unroll
            for (int jj = 0; jj < 4; ++jj) {
                v4bf k4 = cvt4(*(const v4f*)(kp + jj * 4));
                *(v4bf*)&ldsK[key][d0 + jj * 4] = k4;
                v4bf v4 = cvt4(*(const v4f*)(vp + jj * 4));
                #pragma unroll
                for (int e = 0; e < 4; ++e) ldsVT[d0 + jj * 4 + e][key] = v4[e];
            }
        }
        __syncthreads();

        // scores S = Q @ K^T : two 16-col tiles, K-dim 64 = 2 wmma steps
        v8f sc[2];
        for (int t2 = 0; t2 < 2; ++t2)
            for (int r = 0; r < 8; ++r) sc[t2][r] = 0.0f;
        #pragma unroll
        for (int t2 = 0; t2 < 2; ++t2) {
            const int n = t2 * 16 + lm;   // key column in chunk
            #pragma unroll
            for (int s = 0; s < 2; ++s) {
                v16bf bK = ld_frag(&ldsK[n][s * 32 + lh * 8],
                                   &ldsK[n][s * 32 + 16 + lh * 8]);
                sc[t2] = __builtin_amdgcn_wmma_f32_16x16x32_bf16(
                    false, aQ[s], false, bK, (short)0, sc[t2], false, false);
            }
        }

        // causal mask + scale + online softmax (reduce within 16-lane halves)
        #pragma unroll
        for (int r = 0; r < 8; ++r) {
            const int row = qrow0 + lh * 8 + r;
            float v0 = sc[0][r] * SCALEf + ((kb + lm)      > row ? -1e9f : 0.0f);
            float v1 = sc[1][r] * SCALEf + ((kb + 16 + lm) > row ? -1e9f : 0.0f);
            float mx = fmaxf(v0, v1);
            for (int m = 1; m < 16; m <<= 1) mx = fmaxf(mx, __shfl_xor(mx, m));
            float mnew  = fmaxf(rmax[r], mx);
            float scale = __expf(rmax[r] - mnew);
            float p0 = __expf(v0 - mnew);
            float p1 = __expf(v1 - mnew);
            float ls = p0 + p1;
            for (int m = 1; m < 16; m <<= 1) ls += __shfl_xor(ls, m);
            rsum[r] = rsum[r] * scale + ls;
            rmax[r] = mnew;
            for (int t2 = 0; t2 < 4; ++t2) oacc[t2][r] *= scale;
            // bounce P (C-layout) through per-wave LDS to get A-layout
            ldsP[w][lh * 8 + r][lm]      = f2bf(p0);
            ldsP[w][lh * 8 + r][16 + lm] = f2bf(p1);
        }

        v16bf aP = ld_frag(&ldsP[w][lm][lh * 8], &ldsP[w][lm][16 + lh * 8]);
        #pragma unroll
        for (int t2 = 0; t2 < 4; ++t2) {
            v16bf bV = ld_frag(&ldsVT[t2 * 16 + lm][lh * 8],
                               &ldsVT[t2 * 16 + lm][16 + lh * 8]);
            oacc[t2] = __builtin_amdgcn_wmma_f32_16x16x32_bf16(
                false, aP, false, bV, (short)0, oacc[t2], false, false);
        }
        __syncthreads();
    }

    // epilogue: normalize rows, write [B,S,H,D]
    for (int r = 0; r < 8; ++r) {
        const int row = qrow0 + lh * 8 + r;
        const float inv = 1.0f / rsum[r];
        float* op = o + (((size_t)b * Sc + row) * Hc + h) * Dc;
        for (int t2 = 0; t2 < 4; ++t2) op[t2 * 16 + lm] = oacc[t2][r] * inv;
    }
}

// ---------------------------------------------------------------------------
// SwiGLU: m = silu(gate) * up
// ---------------------------------------------------------------------------
__global__ __launch_bounds__(256)
void swiglu_kernel(const float* __restrict__ g, const float* __restrict__ u,
                   float* __restrict__ m, int n) {
    const int i = blockIdx.x * 256 + threadIdx.x;
    if (i < n) {
        const float gv = g[i];
        m[i] = (gv / (1.0f + __expf(-gv))) * u[i];
    }
}

// ---------------------------------------------------------------------------
// Host-side launcher
// ---------------------------------------------------------------------------
extern "C" void kernel_launch(void* const* d_in, const int* in_sizes, int n_in,
                              void* d_out, int out_size, void* d_ws, size_t ws_size,
                              hipStream_t stream) {
    (void)in_sizes; (void)n_in; (void)out_size; (void)ws_size;
    const float* x       = (const float*)d_in[0];
    const float* cosv    = (const float*)d_in[1];
    const float* sinv    = (const float*)d_in[2];
    /* causal_mask d_in[3] applied analytically */
    const float* wq      = (const float*)d_in[4];
    const float* wk      = (const float*)d_in[5];
    const float* wv      = (const float*)d_in[6];
    const float* wo      = (const float*)d_in[7];
    const float* w_gate  = (const float*)d_in[8];
    const float* w_up    = (const float*)d_in[9];
    const float* w_down  = (const float*)d_in[10];
    const float* norm1_w = (const float*)d_in[11];
    const float* norm2_w = (const float*)d_in[12];
    const float* ls_attn = (const float*)d_in[13];
    const float* ls_mlp  = (const float*)d_in[14];
    float* out = (float*)d_out;

    char* ws = (char*)d_ws;
    const size_t SZ_HID = (size_t)Bc * Sc * HIDc * sizeof(float);    // 8 MB
    const size_t SZ_HD  = (size_t)Bc * Sc * Hc * Dc * sizeof(float); // 16 MB
    float* h1   = (float*)(ws);
    float* qb   = (float*)(ws + SZ_HID);
    float* kb   = (float*)(ws + SZ_HID + SZ_HD);
    float* vb   = (float*)(ws + SZ_HID + 2 * SZ_HD);
    float* at   = (float*)(ws + SZ_HID + 3 * SZ_HD);
    float* h2   = (float*)(ws + SZ_HID + 4 * SZ_HD);
    float* gbuf = (float*)(ws + 2 * SZ_HID + 4 * SZ_HD);
    // q/k/v scratch is dead after attention -> reuse for gate/up/m
    float* gate = qb;
    float* up   = kb;
    float* mbuf = vb;

    const int Mrows = Bc * Sc;               // 4096

    // 1. h1 = rmsnorm(x) * norm1_w
    rmsnorm_kernel<<<Mrows, 256, 0, stream>>>(x, norm1_w, h1);
    // 2. QKV projections (K=512, N=1024)
    gemm_bf16_kernel<<<dim3(8, 32), 256, 0, stream>>>(h1, wq, qb, nullptr, nullptr, HIDc, Hc * Dc);
    gemm_bf16_kernel<<<dim3(8, 32), 256, 0, stream>>>(h1, wk, kb, nullptr, nullptr, HIDc, Hc * Dc);
    gemm_bf16_kernel<<<dim3(8, 32), 256, 0, stream>>>(h1, wv, vb, nullptr, nullptr, HIDc, Hc * Dc);
    // 3. RoPE on q,k
    rope_kernel<<<(Bc * Sc * Hc * 32) / 256, 256, 0, stream>>>(qb, kb, cosv, sinv);
    // 4. causal flash attention
    attn_kernel<<<dim3(Bc * Hc, Sc / 64), 128, 0, stream>>>(qb, kb, vb, at);
    // 5. h2 = x + ls_attn * (attn @ wo)   (K=1024, N=512)
    gemm_bf16_kernel<<<dim3(4, 32), 256, 0, stream>>>(at, wo, h2, x, ls_attn, Hc * Dc, HIDc);
    // 6. g = rmsnorm(h2) * norm2_w
    rmsnorm_kernel<<<Mrows, 256, 0, stream>>>(h2, norm2_w, gbuf);
    // 7. gate/up projections (K=512, N=1024)
    gemm_bf16_kernel<<<dim3(8, 32), 256, 0, stream>>>(gbuf, w_gate, gate, nullptr, nullptr, HIDc, INTERc);
    gemm_bf16_kernel<<<dim3(8, 32), 256, 0, stream>>>(gbuf, w_up,   up,   nullptr, nullptr, HIDc, INTERc);
    // 8. m = silu(gate) * up
    swiglu_kernel<<<(Mrows * INTERc) / 256, 256, 0, stream>>>(gate, up, mbuf, Mrows * INTERc);
    // 9. out = h2 + ls_mlp * (m @ w_down)   (K=1024, N=512)
    gemm_bf16_kernel<<<dim3(4, 32), 256, 0, stream>>>(mbuf, w_down, out, h2, ls_mlp, INTERc, HIDc);
}
